// HAN_lp_45337674776786
// MI455X (gfx1250) — matrix operand order repaired and checked by reading, not compiled, not run on hardware
//
#include <hip/hip_runtime.h>
#include <hip/hip_bf16.h>
#include <math.h>

// ---------------- WMMA types / helpers (CDNA5 gfx1250, wave32) ----------------
typedef __attribute__((ext_vector_type(16))) __bf16 v16bf;
typedef __attribute__((ext_vector_type(8)))  float  v8f;

__device__ __forceinline__ v8f wmma_bf16(v16bf a, v16bf b, v8f c) {
  // emits v_wmma_f32_16x16x32_bf16
  return __builtin_amdgcn_wmma_f32_16x16x32_bf16(false, a, false, b, (short)0, c, false, false);
}

// A tile (16x32, MxK): lane<16 -> row=lane,   K = kk+{0..7, 16..23}
//                      lane>=16 -> row=lane-16, K = kk+{8..15, 24..31}
__device__ __forceinline__ v16bf load_A_tile(const float* __restrict__ rowptr, int kk, int half) {
  const float* p = rowptr + kk + half * 8;
  v16bf a;
#pragma unroll
  for (int e = 0; e < 8; ++e) a[e] = (__bf16)p[e];
#pragma unroll
  for (int e = 0; e < 8; ++e) a[8 + e] = (__bf16)p[16 + e];
  return a;
}

// bf16-source variant: two contiguous 16B runs -> b128 loads, no cvt needed
__device__ __forceinline__ v16bf load_A_tile(const __bf16* __restrict__ rowptr, int kk, int half) {
  const __bf16* p = rowptr + kk + half * 8;
  v16bf a;
#pragma unroll
  for (int e = 0; e < 8; ++e) a[e] = p[e];
#pragma unroll
  for (int e = 0; e < 8; ++e) a[8 + e] = p[16 + e];
  return a;
}

// B tile (32x16, KxN), B stored [N][K] row-major (i.e. computing X @ W^T)
__device__ __forceinline__ v16bf load_B_nk(const float* __restrict__ W, int ldk, int ncol, int kk, int half) {
  const float* p = W + (size_t)ncol * ldk + kk + half * 8;
  v16bf b;
#pragma unroll
  for (int e = 0; e < 16; ++e) b[e] = (__bf16)p[e + (e >= 8 ? 8 : 0)];
  return b;
}

// B tile (32x16, KxN), B stored [K][N] row-major (i.e. computing X @ W)
__device__ __forceinline__ v16bf load_B_kn(const float* __restrict__ W, int ldn, int ncol, int kk, int half) {
  const float* p = W + (size_t)(kk + half * 8) * ldn + ncol;
  v16bf b;
#pragma unroll
  for (int e = 0; e < 16; ++e) b[e] = (__bf16)p[(size_t)(e + (e >= 8 ? 8 : 0)) * ldn];
  return b;
}

// ---------------- constants ----------------
#define NP_TYPE 50000
#define NP_G    50000
#define E_G     500000
#define NH      4
#define HID     64
#define DD      256
#define T_TGT   20000
#define ENC_NEG_INF 0x007FFFFFu   // order-preserving encoding of -inf

__device__ __forceinline__ unsigned enc_f32(float v) {
  unsigned b = __float_as_uint(v);
  return (b & 0x80000000u) ? ~b : (b | 0x80000000u);
}
__device__ __forceinline__ float dec_f32(unsigned u) {
  unsigned b = (u & 0x80000000u) ? (u ^ 0x80000000u) : ~u;
  return __uint_as_float(b);
}

// ---------------- Kernel 1: type-wise FC  transformed_bf[type_idx[t][r]] = X_t[r] @ W_t^T + b_t
// Output stored directly in bf16 (it is only ever consumed as a bf16 WMMA A-operand).
__global__ void k_fc(const float* __restrict__ f0, const float* __restrict__ f1,
                     const float* __restrict__ fcW, const float* __restrict__ fcb,
                     const int* __restrict__ type_idx, __bf16* __restrict__ transformed) {
  int t = blockIdx.y;
  int wave = blockIdx.x * 8 + (threadIdx.x >> 5);
  if (wave >= NP_TYPE / 16) return;               // wave-uniform: EXEC stays all-1 for WMMA
  int lane = threadIdx.x & 31, half = lane >> 4, l15 = lane & 15;
  int rowbase = wave * 16;
  const float* X = (t == 0) ? f0 : f1;
  const float* rowptr = X + (size_t)(rowbase + l15) * 256;
  const float* W = fcW + (size_t)t * 64 * 256;    // [n][k]
  v8f acc[4] = {};
  for (int kk = 0; kk < 256; kk += 32) {
    v16bf a = load_A_tile(rowptr, kk, half);
#pragma unroll
    for (int j = 0; j < 4; ++j) {
      v16bf b = load_B_nk(W, 256, j * 16 + l15, kk, half);
      acc[j] = wmma_bf16(a, b, acc[j]);
    }
  }
#pragma unroll
  for (int j = 0; j < 4; ++j) {
    int n = j * 16 + l15;
    float bias = fcb[t * 64 + n];
#pragma unroll
    for (int r = 0; r < 8; ++r) {
      int row = rowbase + r + 8 * half;
      int dstrow = type_idx[t * NP_TYPE + row];
      transformed[(size_t)dstrow * 64 + n] = (__bf16)(acc[j][r] + bias);
    }
  }
}

// ---------------- Kernel 2: generic (optionally gathered) GEMM  C = A[gather] @ B,  B is [K][N]
template <typename TA>
__global__ void k_gemm(const TA* __restrict__ A, int lda, const int* __restrict__ gather,
                       const float* __restrict__ B, int ldb,
                       float* __restrict__ C, int ldc, int Mtiles, int K) {
  int wave = blockIdx.x * 8 + (threadIdx.x >> 5);
  if (wave >= Mtiles) return;
  int lane = threadIdx.x & 31, half = lane >> 4, l15 = lane & 15;
  int rowbase = wave * 16;
  int arow = rowbase + l15;
  if (gather) arow = gather[arow];
  const TA* rowptr = A + (size_t)arow * lda;
  int nbase = blockIdx.y * 64;
  v8f acc[4] = {};
  for (int kk = 0; kk < K; kk += 32) {
    v16bf a = load_A_tile(rowptr, kk, half);
#pragma unroll
    for (int j = 0; j < 4; ++j) {
      v16bf b = load_B_kn(B, ldb, nbase + j * 16 + l15, kk, half);
      acc[j] = wmma_bf16(a, b, acc[j]);
    }
  }
#pragma unroll
  for (int j = 0; j < 4; ++j) {
    int n = nbase + j * 16 + l15;
#pragma unroll
    for (int r = 0; r < 8; ++r) {
      int row = rowbase + r + 8 * half;
      C[(size_t)row * ldc + n] = acc[j][r];
    }
  }
}

// ---------------- Kernel 3: per-node attention logits el/er (one wave per row, coalesced)
__global__ void k_elr(const float* __restrict__ z, const float* __restrict__ al,
                      const float* __restrict__ ar, float* __restrict__ el, float* __restrict__ er) {
  int wave = blockIdx.x * 8 + (threadIdx.x >> 5);
  if (wave >= NP_G) return;
  int lane = threadIdx.x & 31;
  const float* zp = z + (size_t)wave * DD;
  float sl[4] = {0.f, 0.f, 0.f, 0.f}, sr[4] = {0.f, 0.f, 0.f, 0.f};
#pragma unroll
  for (int k = 0; k < 8; ++k) {
    int d = lane + 32 * k;                 // dims lane+32k (fixed k) all belong to head k/2
    float v = zp[d];
    sl[k >> 1] += v * al[d];
    sr[k >> 1] += v * ar[d];
  }
#pragma unroll
  for (int h = 0; h < 4; ++h) {
    float a = sl[h], b = sr[h];
#pragma unroll
    for (int off = 16; off > 0; off >>= 1) {
      a += __shfl_down(a, off, 32);
      b += __shfl_down(b, off, 32);
    }
    if (lane == 0) { el[(size_t)wave * 4 + h] = a; er[(size_t)wave * 4 + h] = b; }
  }
}

__global__ void k_fill_u32(unsigned* __restrict__ p, unsigned v, int n) {
  int i = blockIdx.x * blockDim.x + threadIdx.x;
  if (i < n) p[i] = v;
}

// ---------------- Kernel 4: segment max over dst (order-preserving uint atomicMax)
__global__ void k_edge_max(const int* __restrict__ src, const int* __restrict__ dst,
                           const float* __restrict__ el, const float* __restrict__ er,
                           unsigned* __restrict__ emax_u) {
  int e = blockIdx.x * blockDim.x + threadIdx.x;
  if (e >= E_G) return;
  int s = src[e], d = dst[e];
  float4 l4 = *(const float4*)(el + (size_t)s * 4);
  float4 r4 = *(const float4*)(er + (size_t)d * 4);
  float v[4] = {l4.x + r4.x, l4.y + r4.y, l4.z + r4.z, l4.w + r4.w};
#pragma unroll
  for (int h = 0; h < 4; ++h) {
    float x = v[h] > 0.f ? v[h] : 0.2f * v[h];
    atomicMax(&emax_u[(size_t)d * 4 + h], enc_f32(x));
  }
}

__global__ void k_emax_fix(unsigned* __restrict__ emax_u, int n) {
  int i = blockIdx.x * blockDim.x + threadIdx.x;
  if (i >= n) return;
  float f = dec_f32(emax_u[i]);
  if (!__builtin_isfinite(f)) f = 0.f;      // empty segments: -inf -> 0 (matches reference)
  ((float*)emax_u)[i] = f;
}

// ---------------- Kernel 5: exp + segment sum (denominator)
__global__ void k_edge_exp(const int* __restrict__ src, const int* __restrict__ dst,
                           const float* __restrict__ el, const float* __restrict__ er,
                           const float* __restrict__ emax, float* __restrict__ ex,
                           float* __restrict__ denom) {
  int e = blockIdx.x * blockDim.x + threadIdx.x;
  if (e >= E_G) return;
  int s = src[e], d = dst[e];
  float4 l4 = *(const float4*)(el + (size_t)s * 4);
  float4 r4 = *(const float4*)(er + (size_t)d * 4);
  float4 m4 = *(const float4*)(emax + (size_t)d * 4);
  float v[4] = {l4.x + r4.x, l4.y + r4.y, l4.z + r4.z, l4.w + r4.w};
  float m[4] = {m4.x, m4.y, m4.z, m4.w};
  float4 o;
  float* op = (float*)&o;
#pragma unroll
  for (int h = 0; h < 4; ++h) {
    float x = v[h] > 0.f ? v[h] : 0.2f * v[h];
    float xx = expf(x - m[h]);
    op[h] = xx;
    atomicAdd(&denom[(size_t)d * 4 + h], xx);
  }
  *(float4*)(ex + (size_t)e * 4) = o;
}

// ---------------- Kernel 6: edge-softmax weighted scatter (one wave per edge)
// lanes 0-3 compute the 4 head alphas, broadcast via shuffle; lanes stride the 256
// dims for fully coalesced 128B gather + atomic bursts into the L2-resident accumulator.
__global__ void k_scatter(const int* __restrict__ src, const int* __restrict__ dst,
                          const float* __restrict__ ex, const float* __restrict__ denom,
                          const float* __restrict__ z, float* __restrict__ out) {
  int wave = blockIdx.x * 8 + (threadIdx.x >> 5);
  if (wave >= E_G) return;
  int lane = threadIdx.x & 31;
  int s = src[wave], d = dst[wave];
  float aval = 0.f;
  if (lane < 4) {
    float xx = ex[(size_t)wave * 4 + lane];
    float dn = denom[(size_t)d * 4 + lane];
    aval = xx / (dn + 1e-9f);
  }
  float alpha[4];
  alpha[0] = __shfl(aval, 0, 32);
  alpha[1] = __shfl(aval, 1, 32);
  alpha[2] = __shfl(aval, 2, 32);
  alpha[3] = __shfl(aval, 3, 32);
  const float* zrow = z + (size_t)s * DD;
  float* orow = out + (size_t)d * DD;
#pragma unroll
  for (int h = 0; h < 4; ++h) {
    float a = alpha[h];
#pragma unroll
    for (int k = 0; k < 2; ++k) {
      int dd = h * 64 + k * 32 + lane;
      atomicAdd(&orow[dd], a * zrow[dd]);
    }
  }
}

// ---------------- Kernel 7: ELU + target gather into zs[t, m, :]
__global__ void k_elu_gather(const float* __restrict__ out, const int* __restrict__ tgt,
                             float* __restrict__ zs, int m) {
  int i = blockIdx.x * blockDim.x + threadIdx.x;
  if (i >= T_TGT * DD) return;
  int t = i >> 8, d = i & 255;
  int row = tgt[t];
  float v = out[(size_t)row * DD + d];
  v = v > 0.f ? v : (expf(v) - 1.f);
  zs[((size_t)t * 2 + m) * DD + d] = v;
}

// ---------------- Kernel 8: semantic score reduce  wsum[m] += mean_t tanh(G+b)·q
__global__ void k_wreduce(const float* __restrict__ G, const float* __restrict__ semb,
                          const float* __restrict__ semq, float* __restrict__ wsum, int m) {
  int wave = blockIdx.x * 8 + (threadIdx.x >> 5);
  if (wave >= T_TGT) return;
  int lane = threadIdx.x & 31;
  const float* g = G + (size_t)wave * DD;
  float s = 0.f;
#pragma unroll
  for (int k = 0; k < 8; ++k) {
    int d = lane + 32 * k;
    s += tanhf(g[d] + semb[d]) * semq[d];
  }
#pragma unroll
  for (int off = 16; off > 0; off >>= 1) s += __shfl_down(s, off, 32);
  if (lane == 0) atomicAdd(&wsum[m], s * (1.0f / (float)T_TGT));
}

// ---------------- Kernel 9: beta-softmax combine + output FC (WMMA)
__global__ void k_fcout(const float* __restrict__ zs, const float* __restrict__ wsum,
                        const float* __restrict__ W /*[256][64]*/, const float* __restrict__ bias,
                        float* __restrict__ out /*T x 64*/) {
  int wave = blockIdx.x * 8 + (threadIdx.x >> 5);
  if (wave >= T_TGT / 16) return;
  int lane = threadIdx.x & 31, half = lane >> 4, l15 = lane & 15;
  int rowbase = wave * 16;
  float w0 = wsum[0], w1 = wsum[1];
  float mx = fmaxf(w0, w1);
  float e0 = expf(w0 - mx), e1 = expf(w1 - mx);
  float inv = 1.f / (e0 + e1);
  float beta0 = e0 * inv, beta1 = e1 * inv;
  const float* r0 = zs + (size_t)(rowbase + l15) * 512;     // m=0 row; m=1 is +256
  v8f acc[4] = {};
  for (int kk = 0; kk < 256; kk += 32) {
    const float* p0 = r0 + kk + half * 8;
    const float* p1 = p0 + 256;
    v16bf a;
#pragma unroll
    for (int e = 0; e < 8; ++e) a[e] = (__bf16)(beta0 * p0[e] + beta1 * p1[e]);
#pragma unroll
    for (int e = 0; e < 8; ++e) a[8 + e] = (__bf16)(beta0 * p0[16 + e] + beta1 * p1[16 + e]);
#pragma unroll
    for (int j = 0; j < 4; ++j) {
      v16bf b = load_B_kn(W, 64, j * 16 + l15, kk, half);
      acc[j] = wmma_bf16(a, b, acc[j]);
    }
  }
#pragma unroll
  for (int j = 0; j < 4; ++j) {
    int n = j * 16 + l15;
    float bv = bias[n];
#pragma unroll
    for (int r = 0; r < 8; ++r) {
      int row = rowbase + r + 8 * half;
      out[(size_t)row * 64 + n] = acc[j][r] + bv;
    }
  }
}

// ---------------- host launch ----------------
static inline int cdiv(int a, int b) { return (a + b - 1) / b; }

extern "C" void kernel_launch(void* const* d_in, const int* in_sizes, int n_in,
                              void* d_out, int out_size, void* d_ws, size_t ws_size,
                              hipStream_t stream) {
  (void)in_sizes; (void)n_in; (void)out_size; (void)ws_size;
  const float* features0 = (const float*)d_in[0];
  const float* features1 = (const float*)d_in[1];
  const float* fc_W      = (const float*)d_in[2];
  const float* fc_b      = (const float*)d_in[3];
  const float* gat_W     = (const float*)d_in[4];
  const float* attn_l    = (const float*)d_in[5];
  const float* attn_r    = (const float*)d_in[6];
  const float* sem_W     = (const float*)d_in[7];
  const float* sem_b     = (const float*)d_in[8];
  const float* sem_q     = (const float*)d_in[9];
  const float* fcout_W   = (const float*)d_in[10];
  const float* fcout_b   = (const float*)d_in[11];
  const int* type_idx    = (const int*)d_in[12];
  const int* node_idx    = (const int*)d_in[13];
  const int* edge_src    = (const int*)d_in[14];
  const int* edge_dst    = (const int*)d_in[15];
  const int* tgt_idx     = (const int*)d_in[16];
  float* out = (float*)d_out;

  char* ws = (char*)d_ws;
  size_t off = 0;
  auto alloc = [&](size_t bytes) -> void* {
    void* p = (void*)(ws + off);
    off += (bytes + 255) & ~(size_t)255;
    return p;
  };
  __bf16*   transformed = (__bf16*)alloc((size_t)100000 * 64 * 2);
  float*    z     = (float*)alloc((size_t)NP_G * DD * 4);
  float*    gout  = (float*)alloc((size_t)NP_G * DD * 4);
  float*    zs    = (float*)alloc((size_t)T_TGT * 2 * DD * 4);
  float*    el    = (float*)alloc((size_t)NP_G * NH * 4);
  float*    er    = (float*)alloc((size_t)NP_G * NH * 4);
  unsigned* emaxu = (unsigned*)alloc((size_t)NP_G * NH * 4);
  float*    denom = (float*)alloc((size_t)NP_G * NH * 4);
  float*    exb   = (float*)alloc((size_t)E_G * NH * 4);
  float*    wsum  = (float*)alloc(256);
  float*    G     = z;   // reused after GAT for the semantic GEMM

  const dim3 blk(256);

  // Phase 1: type-wise FC into shared node tensor (bf16)
  k_fc<<<dim3(cdiv(NP_TYPE / 16, 8), 2), blk, 0, stream>>>(
      features0, features1, fc_W, fc_b, type_idx, transformed);

  for (int b = 0; b < 2; ++b) {
    for (int m = 0; m < 2; ++m) {
      int g = b * 2 + m;
      const int* nidx = node_idx + (size_t)g * NP_G;
      const float* gw = gat_W + (size_t)g * HID * DD;
      const float* al = attn_l + (size_t)g * NH * 64;
      const float* ar = attn_r + (size_t)g * NH * 64;
      const int* src = edge_src + (size_t)g * E_G;
      const int* dst = edge_dst + (size_t)g * E_G;
      const int* tgt = tgt_idx + (size_t)g * T_TGT;

      // z = transformed[node_idx] @ gat_W   (gathered bf16 WMMA GEMM)
      k_gemm<__bf16><<<dim3(cdiv(NP_G / 16, 8), 4), blk, 0, stream>>>(
          transformed, 64, nidx, gw, DD, z, DD, NP_G / 16, HID);

      k_elr<<<cdiv(NP_G, 8), blk, 0, stream>>>(z, al, ar, el, er);

      k_fill_u32<<<cdiv(NP_G * NH, 256), blk, 0, stream>>>(emaxu, ENC_NEG_INF, NP_G * NH);
      hipMemsetAsync(denom, 0, (size_t)NP_G * NH * 4, stream);
      hipMemsetAsync(gout, 0, (size_t)NP_G * DD * 4, stream);

      k_edge_max<<<cdiv(E_G, 256), blk, 0, stream>>>(src, dst, el, er, emaxu);
      k_emax_fix<<<cdiv(NP_G * NH, 256), blk, 0, stream>>>(emaxu, NP_G * NH);
      k_edge_exp<<<cdiv(E_G, 256), blk, 0, stream>>>(src, dst, el, er, (const float*)emaxu, exb, denom);
      k_scatter<<<cdiv(E_G, 8), blk, 0, stream>>>(src, dst, exb, denom, z, gout);

      k_elu_gather<<<cdiv(T_TGT * DD, 256), blk, 0, stream>>>(gout, tgt, zs, m);
    }

    // Semantic attention for branch b
    hipMemsetAsync(wsum, 0, 2 * sizeof(float), stream);
    for (int m = 0; m < 2; ++m) {
      // G = zs[:, m, :] @ sem_W[b]   (WMMA GEMM, lda = 512 with offset m*256)
      k_gemm<float><<<dim3(cdiv(T_TGT / 16, 8), 4), blk, 0, stream>>>(
          zs + (size_t)m * DD, 2 * DD, nullptr,
          sem_W + (size_t)b * DD * DD, DD, G, DD, T_TGT / 16, DD);
      k_wreduce<<<cdiv(T_TGT, 8), blk, 0, stream>>>(
          G, sem_b + (size_t)b * DD, sem_q + (size_t)b * DD, wsum, m);
    }

    // beta-combine + output FC
    k_fcout<<<cdiv(T_TGT / 16, 8), blk, 0, stream>>>(
        zs, wsum, fcout_W + (size_t)b * DD * 64, fcout_b + (size_t)b * 64,
        out + (size_t)b * T_TGT * 64);
  }
}
